// MultiHeadAttention_12773232738828
// MI455X (gfx1250) — matrix operand order repaired and checked by reading, not compiled, not run on hardware
//
#include <hip/hip_runtime.h>
#include <hip/hip_bf16.h>

typedef __bf16 bf16_t;
typedef __attribute__((ext_vector_type(16))) __bf16 v16bf;
typedef __attribute__((ext_vector_type(8)))  __bf16 v8bf;
typedef __attribute__((ext_vector_type(8)))  float  v8f;

constexpr int B_ = 2, S_ = 2048, D_ = 1024, H_ = 16, HD_ = 64;
constexpr float SCALE_ = 0.125f;   // 64^-0.5  (folded into Q projection)

#if defined(__has_builtin)
#if __has_builtin(__builtin_amdgcn_global_load_async_to_lds_b128)
#define HAVE_ASYNC 1
#endif
#endif
#ifndef HAVE_ASYNC
#define HAVE_ASYNC 0
#endif

#if HAVE_ASYNC
typedef int async_v4i __attribute__((vector_size(16)));
typedef __attribute__((address_space(1))) async_v4i* g_b128_t;
typedef __attribute__((address_space(3))) async_v4i* l_b128_t;
#endif

__device__ __forceinline__ v16bf cat8(v8bf lo, v8bf hi) {
  return __builtin_shufflevector(lo, hi, 0,1,2,3,4,5,6,7,8,9,10,11,12,13,14,15);
}
__device__ __forceinline__ v8f wmma_bf16(v16bf a, v16bf b, v8f c) {
  return __builtin_amdgcn_wmma_f32_16x16x32_bf16(false, a, false, b, (short)0, c,
                                                 false, false);
}

// ---- 16-lane (half-wave) max butterfly in pure VALU via DPP ----------------
template <int CTRL>
__device__ __forceinline__ float dppmax(float v) {
  int s = __builtin_amdgcn_update_dpp(0, __float_as_int(v), CTRL, 0xF, 0xF, true);
  return fmaxf(v, __int_as_float(s));
}
__device__ __forceinline__ float redmax16(float v) {
  v = dppmax<0xB1>(v);  v = dppmax<0x4E>(v);   // quad_perm xor1, xor2
  v = dppmax<0x141>(v); v = dppmax<0x140>(v);  // row_half_mirror, row_mirror
  return v;
}

// copy 16 bytes global -> LDS; async (no VGPR data path, ASYNCcnt) if available
__device__ __forceinline__ void copy16(const bf16_t* g, bf16_t* l) {
#if HAVE_ASYNC
  __builtin_amdgcn_global_load_async_to_lds_b128(
      (g_b128_t)(void*)const_cast<bf16_t*>(g), (l_b128_t)(void*)l, 0, 0);
#else
  *(v8bf*)l = *(const v8bf*)g;
#endif
}
__device__ __forceinline__ void wait_async() {
#if HAVE_ASYNC
#if defined(__has_builtin) && __has_builtin(__builtin_amdgcn_s_wait_asynccnt)
  __builtin_amdgcn_s_wait_asynccnt(0);
#else
  asm volatile("s_wait_asynccnt 0" ::: "memory");
#endif
#endif
}

// ---------------------------------------------------------------------------
// GEMM: Y[m,n] = (sum_k A[m,k] * W[n,k] + bias[n]) * oscale
// OUT_MODE 0: bf16 head-split   [B,H,S,HD]   (Q with oscale=SCALE, K)
// OUT_MODE 1: bf16 head-split-T [B,H,HD,S]   (V; PV B-frag becomes contiguous)
// OUT_MODE 2: f32 plain         [M,N]        (final output projection)
// Block tile 128x128, 8 waves (2x4), each wave 64x32 = 4x2 WMMA tiles, BK=32.
// Double-buffered LDS: one barrier per K-step.
// ---------------------------------------------------------------------------
template <int OUT_MODE, typename AT>
__global__ __launch_bounds__(256) void gemm_proj(const AT* __restrict__ A,
                                                 const float* __restrict__ W,
                                                 const float* __restrict__ bias,
                                                 void* __restrict__ outp,
                                                 float oscale) {
  constexpr int K = D_, BM = 128, BN = 128, BK = 32, LD = 40;  // 80B row pitch
  __shared__ bf16_t As[2][BM][LD];
  __shared__ bf16_t Ws[2][BN][LD];

  const int tid  = threadIdx.x;
  const int wave = tid >> 5, lane = tid & 31;
  const int wm = wave & 1, wn = wave >> 1;  // 2 (M) x 4 (N) waves
  const int hf = lane >> 4, ln = lane & 15;
  const int m0 = blockIdx.x * BM, n0 = blockIdx.y * BN;

  auto loadTiles = [&](int k0, int buf) {
    #pragma unroll
    for (int i = tid; i < BM * BK; i += 256) {
      int r = i >> 5, c = i & 31;
      As[buf][r][c] = (bf16_t)A[(size_t)(m0 + r) * K + k0 + c];
    }
    #pragma unroll
    for (int i = tid; i < BN * BK; i += 256) {
      int r = i >> 5, c = i & 31;
      Ws[buf][r][c] = (bf16_t)W[(size_t)(n0 + r) * K + k0 + c];
    }
  };

  v8f acc[4][2] = {};
  loadTiles(0, 0);
  __syncthreads();

  int par = 0;
  for (int k0 = 0; k0 < K; k0 += BK) {
    if (k0 + BK < K) loadTiles(k0 + BK, par ^ 1);  // prefetch next tile

    v16bf afr[4], bfr[2];
    #pragma unroll
    for (int tm = 0; tm < 4; ++tm) {  // A 16x32: K = hf*8+j | 16+hf*8+j
      const bf16_t* p = &As[par][wm * 64 + tm * 16 + ln][hf * 8];
      afr[tm] = cat8(*(const v8bf*)p, *(const v8bf*)(p + 16));
    }
    #pragma unroll
    for (int tn = 0; tn < 2; ++tn) {  // B 32x16: K = hf*8+j | 16+hf*8+j
      const bf16_t* p = &Ws[par][wn * 32 + tn * 16 + ln][hf * 8];
      bfr[tn] = cat8(*(const v8bf*)p, *(const v8bf*)(p + 16));
    }
    #pragma unroll
    for (int tm = 0; tm < 4; ++tm)
      #pragma unroll
      for (int tn = 0; tn < 2; ++tn)
        acc[tm][tn] = wmma_bf16(afr[tm], bfr[tn], acc[tm][tn]);

    __syncthreads();
    par ^= 1;
  }

  // epilogue: C layout is (M = r + 8*hf, N = ln) per VGPR r
  #pragma unroll
  for (int tm = 0; tm < 4; ++tm)
    #pragma unroll
    for (int tn = 0; tn < 2; ++tn)
      #pragma unroll
      for (int r = 0; r < 8; ++r) {
        int m = m0 + wm * 64 + tm * 16 + r + hf * 8;
        int n = n0 + wn * 32 + tn * 16 + ln;
        float val = (acc[tm][tn][r] + bias[n]) * oscale;
        if constexpr (OUT_MODE == 0) {
          int bb = m >> 11, ss = m & (S_ - 1);
          int hh = n >> 6,  hd = n & (HD_ - 1);
          ((bf16_t*)outp)[(((size_t)(bb * H_ + hh)) * S_ + ss) * HD_ + hd] =
              (bf16_t)val;
        } else if constexpr (OUT_MODE == 1) {
          int bb = m >> 11, ss = m & (S_ - 1);
          int hh = n >> 6,  hd = n & (HD_ - 1);
          ((bf16_t*)outp)[(((size_t)(bb * H_ + hh)) * HD_ + hd) * S_ + ss] =
              (bf16_t)val;
        } else {
          ((float*)outp)[(size_t)m * D_ + n] = val;
        }
      }
}

// ---------------------------------------------------------------------------
// Flash attention. 256-thread block = 8 waves = 128 query rows of one (b,h);
// all 8 waves share K/V tiles (KV block = 32) staged in double-buffered LDS
// via async global->LDS copies (ASYNCcnt). Online softmax: max via DPP
// butterfly, row-sum via an extra WMMA against an all-ones B fragment.
// Q,K: [B,H,S,HD] bf16.  V: [B,H,HD,S] bf16.  Out: token-major bf16 [B,S,D].
// ---------------------------------------------------------------------------
__global__ __launch_bounds__(256) void attn_kernel(const bf16_t* __restrict__ Qh,
                                                   const bf16_t* __restrict__ Kh,
                                                   const bf16_t* __restrict__ Vt,
                                                   const int* __restrict__ mask,
                                                   bf16_t* __restrict__ Obuf) {
  __shared__ bf16_t Ks[2][32][72];  // [kv][hd],  pitch 144B (bank-spread)
  __shared__ bf16_t Vs[2][64][40];  // [hd][kv],  pitch  80B
  __shared__ bf16_t Pl[8][16][40];  // per-wave P staging (C-layout->A-layout)

  const int tid = threadIdx.x;
  const int lane = tid & 31, wave = tid >> 5;
  const int hf = lane >> 4, ln = lane & 15;
  const int q0 = (blockIdx.x * 8 + wave) * 16;
  const int h = blockIdx.y, b = blockIdx.z;
  const int bh = b * H_ + h;

  const bf16_t* Qp = Qh + (size_t)bh * S_ * HD_;
  const bf16_t* Kp = Kh + (size_t)bh * S_ * HD_;
  const bf16_t* Vp = Vt + (size_t)bh * HD_ * S_;
  const int*    mp = mask + (size_t)b * S_ * S_;

  // cooperative K/V tile staging: one 16B chunk per thread per matrix
  const int krow = tid >> 3, kch = (tid & 7) * 8;   // 32 rows x 8 chunks(16B)
  const int vrow = tid >> 2, vch = (tid & 3) * 8;   // 64 rows x 4 chunks(16B)
  auto stage = [&](int j0, int buf) {
    copy16(Kp + (size_t)(j0 + krow) * HD_ + kch, &Ks[buf][krow][kch]);
    copy16(Vp + (size_t)vrow * S_ + j0 + vch,    &Vs[buf][vrow][vch]);
  };

  // Q tile 16x64 as two A-fragments (K-chunks of 32), kept in registers
  v16bf qa[2];
  #pragma unroll
  for (int c = 0; c < 2; ++c) {
    const bf16_t* p = Qp + (size_t)(q0 + ln) * HD_ + c * 32 + hf * 8;
    qa[c] = cat8(*(const v8bf*)p, *(const v8bf*)(p + 16));
  }

  // all-ones B fragment: rowsum(P) = P @ ones via one WMMA
  v16bf onesb;
  #pragma unroll
  for (int j = 0; j < 16; ++j) onesb[j] = (bf16_t)1.0f;

  v8f o[4] = {};           // O accumulator 16x64 (4 N-tiles of 16)
  float mrow[8], lrow[8];  // per-row running max / sum (row = r + 8*hf)
  #pragma unroll
  for (int r = 0; r < 8; ++r) { mrow[r] = -3.0e38f; lrow[r] = 0.f; }

  stage(0, 0);
  int par = 0;
  for (int j0 = 0; j0 < S_; j0 += 32) {
    wait_async();          // own async copies for buf[par] complete
    __syncthreads();       // all waves' copies visible; prev readers done
    if (j0 + 32 < S_) stage(j0 + 32, par ^ 1);  // overlap next tile staging

    // ---- S = Q @ K^T : load all 4 K-frags first, then 4 WMMA
    v16bf kb[2][2];
    #pragma unroll
    for (int tn = 0; tn < 2; ++tn)
      #pragma unroll
      for (int c = 0; c < 2; ++c) {
        const bf16_t* p = &Ks[par][tn * 16 + ln][c * 32 + hf * 8];
        kb[tn][c] = cat8(*(const v8bf*)p, *(const v8bf*)(p + 16));
      }
    v8f sc[2] = {};
    #pragma unroll
    for (int tn = 0; tn < 2; ++tn)
      #pragma unroll
      for (int c = 0; c < 2; ++c)
        sc[tn] = wmma_bf16(qa[c], kb[tn][c], sc[tn]);

    // ---- mask (reference: masked -> -1e9 before softmax); Q carries SCALE
    #pragma unroll
    for (int tn = 0; tn < 2; ++tn)
      #pragma unroll
      for (int r = 0; r < 8; ++r) {
        int qr = q0 + r + hf * 8;
        int col = j0 + tn * 16 + ln;
        int mv = mp[(size_t)qr * S_ + col];
        sc[tn][r] = mv ? sc[tn][r] : -1.0e9f;
      }

    // ---- online softmax: row max via DPP butterfly over the half-wave
    float mnew[8], corr[8];
    #pragma unroll
    for (int r = 0; r < 8; ++r) {
      float mx = redmax16(fmaxf(sc[0][r], sc[1][r]));
      mnew[r] = fmaxf(mrow[r], mx);
      corr[r] = __expf(mrow[r] - mnew[r]);
      mrow[r] = mnew[r];
    }
    #pragma unroll
    for (int tn = 0; tn < 2; ++tn)
      #pragma unroll
      for (int r = 0; r < 8; ++r)
        sc[tn][r] = __expf(sc[tn][r] - mnew[r]);
    #pragma unroll
    for (int t = 0; t < 4; ++t)
      #pragma unroll
      for (int r = 0; r < 8; ++r) o[t][r] *= corr[r];

    // ---- relayout P (C-frag f32) -> A-frag bf16 via per-wave LDS slice
    #pragma unroll
    for (int tn = 0; tn < 2; ++tn)
      #pragma unroll
      for (int r = 0; r < 8; ++r)
        Pl[wave][r + hf * 8][tn * 16 + ln] = (bf16_t)sc[tn][r];
    asm volatile("s_wait_dscnt 0" ::: "memory");  // in-wave store->load fence
    const bf16_t* pp = &Pl[wave][ln][hf * 8];
    v16bf pa = cat8(*(const v8bf*)pp, *(const v8bf*)(pp + 16));

    // ---- load all 4 V-frags, then: rowsum WMMA + 4 PV WMMA
    v16bf vb[4];
    #pragma unroll
    for (int t = 0; t < 4; ++t) {
      const bf16_t* vp = &Vs[par][t * 16 + ln][hf * 8];
      vb[t] = cat8(*(const v8bf*)vp, *(const v8bf*)(vp + 16));
    }
    v8f ls = {};
    ls = wmma_bf16(pa, onesb, ls);   // rowsum(P), column-replicated in C
    #pragma unroll
    for (int t = 0; t < 4; ++t) o[t] = wmma_bf16(pa, vb[t], o[t]);
    #pragma unroll
    for (int r = 0; r < 8; ++r) lrow[r] = lrow[r] * corr[r] + ls[r];

    par ^= 1;
  }

  // ---- normalize and store token-major bf16 [B,S,D]
  #pragma unroll
  for (int r = 0; r < 8; ++r) lrow[r] = 1.0f / lrow[r];
  #pragma unroll
  for (int t = 0; t < 4; ++t)
    #pragma unroll
    for (int r = 0; r < 8; ++r) {
      int qr = q0 + r + hf * 8;
      int col = h * HD_ + t * 16 + ln;
      Obuf[(size_t)(b * S_ + qr) * D_ + col] = (bf16_t)(o[t][r] * lrow[r]);
    }
}

// ---------------------------------------------------------------------------
extern "C" void kernel_launch(void* const* d_in, const int* in_sizes, int n_in,
                              void* d_out, int out_size, void* d_ws,
                              size_t ws_size, hipStream_t stream) {
  const float* q    = (const float*)d_in[0];
  const float* k    = (const float*)d_in[1];
  const float* v    = (const float*)d_in[2];
  const int*   mask = (const int*)d_in[3];
  const float* Wq   = (const float*)d_in[4];
  const float* bq   = (const float*)d_in[5];
  const float* Wk   = (const float*)d_in[6];
  const float* bk   = (const float*)d_in[7];
  const float* Wv   = (const float*)d_in[8];
  const float* bv   = (const float*)d_in[9];
  const float* Wo   = (const float*)d_in[10];
  const float* bo   = (const float*)d_in[11];

  const size_t nElem = (size_t)B_ * S_ * D_;  // 4,194,304
  bf16_t* Qh = (bf16_t*)d_ws;                 //  8 MB
  bf16_t* Kh = Qh + nElem;                    //  8 MB
  bf16_t* Vt = Kh + nElem;                    //  8 MB
  bf16_t* Ob = Vt + nElem;                    //  8 MB  (32 MB total)

  dim3 gg(B_ * S_ / 128, D_ / 128);           // 32 x 8 blocks, 256 threads
  gemm_proj<0, float><<<gg, 256, 0, stream>>>(q, Wq, bq, Qh, SCALE_);
  gemm_proj<0, float><<<gg, 256, 0, stream>>>(k, Wk, bk, Kh, 1.0f);
  gemm_proj<1, float><<<gg, 256, 0, stream>>>(v, Wv, bv, Vt, 1.0f);

  attn_kernel<<<dim3(S_ / 128, H_, B_), 256, 0, stream>>>(Qh, Kh, Vt, mask, Ob);

  gemm_proj<2, bf16_t><<<gg, 256, 0, stream>>>(Ob, Wo, bo, d_out, 1.0f);
}
